// CrossModalFFTAttn_V5_50019189129723
// MI455X (gfx1250) — matrix operand order repaired and checked by reading, not compiled, not run on hardware
//
#include <hip/hip_runtime.h>
#include <hip/hip_bf16.h>

// ---------------------------------------------------------------------------
// CDNA5 (gfx1250) CrossModalFFTAttn_V5.
//   rstd_x/rstd_evt -> WMMA GEMM conv1x1 (q,kv)  [async-LDS double buffered]
//   -> dwconv3x3 -> rstd_q/rstd_k -> 8x8 patch circular conv (== FFT corr)
//   -> rstd_corr -> WMMA GEMM proj (v*corr)      [async-LDS double buffered]
// ---------------------------------------------------------------------------

typedef __attribute__((ext_vector_type(16))) __bf16 v16bf;
typedef __attribute__((ext_vector_type(8)))  float  v8f;

#define EPSV 1e-6f

// ---------------- gfx1250 async-copy helpers (ASYNCcnt tracked) ------------
__device__ __forceinline__ void async_b64_to_lds(unsigned lds_off,
                                                 unsigned long long gaddr) {
    // GLOBAL_LOAD_ASYNC_TO_LDS_B64: per-lane 8-byte copy from global address to LDS offset
    asm volatile("global_load_async_to_lds_b64 %0, %1, off"
                 :: "v"(lds_off), "v"(gaddr) : "memory");
}
template <int N>
__device__ __forceinline__ void wait_asynccnt() {
    asm volatile("s_wait_asynccnt %0" :: "i"(N) : "memory");
}

// -------------------------- f32 -> bf16 convert ----------------------------
__global__ void cvt_f32_bf16_kernel(const float* __restrict__ src,
                                    __bf16* __restrict__ dst, int n) {
    int i = blockIdx.x * 256 + threadIdx.x;
    if (i < n) dst[i] = (__bf16)src[i];
}

// ---------------- rms stats over channel dim (f32 input) -------------------
__global__ void rms_stats_f32_kernel(const float* __restrict__ in,
                                     float* __restrict__ rstd,
                                     int C, int HW, int BHW) {
    int idx = blockIdx.x * 256 + threadIdx.x;
    if (idx >= BHW) return;
    int b = idx / HW, pos = idx - b * HW;
    const float* p = in + ((size_t)b * C) * HW + pos;
    float s = 0.f;
    for (int c = 0; c < C; ++c) {
        float v = p[(size_t)c * HW];
        s += v * v;
    }
    rstd[idx] = rsqrtf(s / (float)C + EPSV);
}

// ---------------- rms stats over channel dim (bf16 input) ------------------
__global__ void rms_stats_bf16_kernel(const __bf16* __restrict__ in,
                                      float* __restrict__ rstd,
                                      int Ctot, int choff, int Cnorm,
                                      int HW, int BHW) {
    int idx = blockIdx.x * 256 + threadIdx.x;
    if (idx >= BHW) return;
    int b = idx / HW, pos = idx - b * HW;
    const __bf16* p = in + ((size_t)b * Ctot + choff) * HW + pos;
    float s = 0.f;
    for (int c = 0; c < Cnorm; ++c) {
        float v = (float)p[(size_t)c * HW];
        s += v * v;
    }
    rstd[idx] = rsqrtf(s / (float)Cnorm + EPSV);
}

// ------------- conv1x1 as WMMA GEMM, rmsnorm fused into B-load -------------
// One wave computes a (16*MBLK) x 16 output block.  B tile (32K x 16N f32)
// is async-copied to LDS, double buffered across the unrolled K loop.
// LDS f32 tile row stride = 18 dwords: keeps lane halves (hi=0/1) on
// disjoint bank groups and 8B alignment for b64 async writes.
template <int KK, int MBLK>
__global__ void gemm_conv1x1_bf16_kernel(const float*  __restrict__ in,   // (B,KK,HW)
                                         const __bf16* __restrict__ w16,  // (M,KK)
                                         const float*  __restrict__ wnorm,// (KK)
                                         const float*  __restrict__ rstd, // (B,HW)
                                         __bf16*       __restrict__ out,  // (B,M,HW)
                                         int M, int HW) {
    constexpr int STEPS = KK / 32;
    constexpr int TROW  = 18;               // f32 tile row stride (dwords)
    __shared__ float tile[2][32 * TROW];

    const int lane = threadIdx.x & 31;
    const int nc   = lane & 15;
    const int hi   = lane >> 4;
    const int n0   = blockIdx.x * 16;
    const int m0   = blockIdx.y * (16 * MBLK);
    const int b    = blockIdx.z;

    const int   pos  = n0 + nc;
    const float rinv = rstd[(size_t)b * HW + pos];
    const float* colbase = in + (size_t)b * KK * HW + n0;

    const unsigned ldsbase = (unsigned)(size_t)(&tile[0][0]);
    constexpr unsigned TBYTES = 32u * TROW * 4u;

    // async-copy one 32x16 f32 tile (2KB) = 8 b64 ops per wave
    auto issue_tile = [&](int k0, int buf) {
#pragma unroll
        for (int t = 0; t < 8; ++t) {
            int e   = t * 32 + lane;        // 0..255 : 8B elements
            int row = e >> 3, seg = e & 7;
            unsigned long long g =
                (unsigned long long)(const void*)(colbase + (size_t)(k0 + row) * HW + seg * 2);
            async_b64_to_lds(ldsbase + (unsigned)buf * TBYTES +
                             (unsigned)(row * TROW * 4 + seg * 8), g);
        }
    };

    issue_tile(0, 0);

    v8f acc[MBLK];
#pragma unroll
    for (int mb = 0; mb < MBLK; ++mb) acc[mb] = (v8f){};

#pragma unroll
    for (int s = 0; s < STEPS; ++s) {
        const int k0 = s * 32;
        if (s + 1 < STEPS) {
            issue_tile(k0 + 32, (s + 1) & 1);
            wait_asynccnt<8>();             // current tile done, next in flight
        } else {
            wait_asynccnt<0>();
        }
        // B fragment from LDS: lane holds col nc, K rows k0 + hi*16 + i
        v16bf bb;
        {
            const float* T = &tile[s & 1][0];
#pragma unroll
            for (int i = 0; i < 16; ++i) {
                float v = T[(hi * 16 + i) * TROW + nc] *
                          wnorm[k0 + hi * 16 + i] * rinv;
                bb[i] = (__bf16)v;
            }
        }
        // A fragments (two aligned b128 loads each) + MBLK WMMAs
#pragma unroll
        for (int mb = 0; mb < MBLK; ++mb) {
            union { uint4 u[2]; v16bf v; } A;
            const uint4* wr = (const uint4*)(w16 +
                (size_t)(m0 + mb * 16 + nc) * KK + k0 + hi * 16);
            if (s + 1 < STEPS) __builtin_prefetch(wr + 4, 0, 1);
            A.u[0] = wr[0];
            A.u[1] = wr[1];
            acc[mb] = __builtin_amdgcn_wmma_f32_16x16x32_bf16(
                false, A.v, false, bb, (short)0, acc[mb], false, false);
        }
    }
    __bf16* ob = out + (size_t)b * M * HW;
#pragma unroll
    for (int mb = 0; mb < MBLK; ++mb)
#pragma unroll
        for (int r = 0; r < 8; ++r)
            ob[(size_t)(m0 + mb * 16 + hi * 8 + r) * HW + pos] = (__bf16)acc[mb][r];
}

// ------------------------- depthwise 3x3, SAME -----------------------------
__global__ void dwconv3x3_kernel(const __bf16* __restrict__ in,
                                 const float*  __restrict__ w9,  // (C,9)
                                 __bf16*       __restrict__ out,
                                 int C, int H, int W, int total) {
    int idx = blockIdx.x * 256 + threadIdx.x;
    if (idx >= total) return;
    const int HW = H * W;
    int plane = idx / HW;            // b*C + c
    int rem   = idx - plane * HW;
    int hy = rem / W, wx = rem - hy * W;
    int c = plane % C;
    const float* wk = w9 + (size_t)c * 9;
    const __bf16* ip = in + (size_t)plane * HW;
    float acc = 0.f;
#pragma unroll
    for (int dy = -1; dy <= 1; ++dy) {
        int y = hy + dy;
        if ((unsigned)y >= (unsigned)H) continue;
#pragma unroll
        for (int dx = -1; dx <= 1; ++dx) {
            int x = wx + dx;
            if ((unsigned)x >= (unsigned)W) continue;
            acc += (float)ip[(size_t)y * W + x] * wk[(dy + 1) * 3 + (dx + 1)];
        }
    }
    out[idx] = (__bf16)acc;
}

// --------- 8x8 patch circular convolution (== irfft2(rfft2*rfft2)) ---------
__global__ void patch_corr_kernel(const __bf16* __restrict__ q,    // (B,C2,H,W)
                                  const __bf16* __restrict__ kv,   // (B,2*C2,H,W)
                                  const float*  __restrict__ wqn,
                                  const float*  __restrict__ wkn,
                                  const float*  __restrict__ rstdq,
                                  const float*  __restrict__ rstdk,
                                  __bf16*       __restrict__ corr,
                                  int C2, int H, int W) {
    __shared__ float qs[4][64];
    __shared__ float ks[4][64];
    const int HW = H * W;
    const int PW = W >> 3, PH = H >> 3;
    int blk = blockIdx.x;
    int pw = blk % PW; int t = blk / PW;
    int ph = t % PH;   int b = t / PH;

    const int tid = threadIdx.x;
    const int o   = tid & 63;
    const int cs  = tid >> 6;
    const int ey = o >> 3, ex = o & 7;
    const int hh = ph * 8 + ey, ww = pw * 8 + ex;
    const size_t posIdx = (size_t)hh * W + ww;
    const float rsq = rstdq[(size_t)b * HW + posIdx];
    const float rsk = rstdk[(size_t)b * HW + posIdx];
    const int m = ey, n = ex;

    for (int c0 = 0; c0 < C2; c0 += 4) {
        int c = c0 + cs;
        size_t qidx = ((size_t)b * C2 + c) * HW + posIdx;
        size_t kidx = ((size_t)b * 2 * C2 + c) * HW + posIdx;
        __syncthreads();
        qs[cs][o] = (float)q[qidx]  * wqn[c] * rsq;
        ks[cs][o] = (float)kv[kidx] * wkn[c] * rsk;
        __syncthreads();
        float acc = 0.f;
#pragma unroll
        for (int i = 0; i < 8; ++i) {
            int km = ((m - i) & 7) << 3;
#pragma unroll
            for (int j = 0; j < 8; ++j) {
                acc += qs[cs][(i << 3) + j] * ks[cs][km + ((n - j) & 7)];
            }
        }
        corr[qidx] = (__bf16)acc;
    }
}

// --------- final proj GEMM: out = W_proj x (v * corr * norm), f32 out ------
// v and corr bf16 tiles async-copied to LDS, double buffered.
// bf16 tile row stride = 20 elems (40B): disjoint bank halves, 8B aligned.
template <int KK, int MBLK>
__global__ void gemm_vproj_kernel(const __bf16* __restrict__ kvdw, // (B,512,HW), v = ch [256,512)
                                  const __bf16* __restrict__ corr, // (B,256,HW)
                                  const __bf16* __restrict__ w16,  // (128,KK)
                                  const float*  __restrict__ wnc,  // (KK)
                                  const float*  __restrict__ rstdc,// (B,HW)
                                  float*        __restrict__ out,  // (B,128,HW)
                                  int M, int HW) {
    constexpr int STEPS = KK / 32;
    constexpr int TROW  = 20;               // bf16 tile row stride (elems)
    __shared__ __bf16 vt[2][32 * TROW];
    __shared__ __bf16 ct[2][32 * TROW];

    const int lane = threadIdx.x & 31;
    const int nc = lane & 15;
    const int hi = lane >> 4;
    const int n0 = blockIdx.x * 16;
    const int m0 = blockIdx.y * (16 * MBLK);
    const int b  = blockIdx.z;

    const int   pos  = n0 + nc;
    const float rinv = rstdc[(size_t)b * HW + pos];
    const __bf16* vb = kvdw + ((size_t)b * 512 + 256) * HW + n0;
    const __bf16* cb = corr + (size_t)b * 256 * HW + n0;

    const unsigned vbase = (unsigned)(size_t)(&vt[0][0]);
    const unsigned cbase = (unsigned)(size_t)(&ct[0][0]);
    constexpr unsigned TBYTES = 32u * TROW * 2u;

    // one 32x16 bf16 tile (1KB) = 4 b64 ops per tensor
    auto issue_tiles = [&](int k0, int buf) {
#pragma unroll
        for (int t = 0; t < 4; ++t) {
            int e   = t * 32 + lane;        // 0..127 : 8B elements
            int row = e >> 2, seg = e & 3;
            unsigned loff = (unsigned)buf * TBYTES +
                            (unsigned)(row * TROW * 2 + seg * 8);
            async_b64_to_lds(vbase + loff,
                (unsigned long long)(const void*)(vb + (size_t)(k0 + row) * HW + seg * 4));
            async_b64_to_lds(cbase + loff,
                (unsigned long long)(const void*)(cb + (size_t)(k0 + row) * HW + seg * 4));
        }
    };

    issue_tiles(0, 0);

    v8f acc[MBLK];
#pragma unroll
    for (int mb = 0; mb < MBLK; ++mb) acc[mb] = (v8f){};

#pragma unroll
    for (int s = 0; s < STEPS; ++s) {
        const int k0 = s * 32;
        if (s + 1 < STEPS) {
            issue_tiles(k0 + 32, (s + 1) & 1);
            wait_asynccnt<8>();
        } else {
            wait_asynccnt<0>();
        }
        v16bf bb;
        {
            const __bf16* Tv = &vt[s & 1][0];
            const __bf16* Tc = &ct[s & 1][0];
#pragma unroll
            for (int i = 0; i < 16; ++i) {
                int r = (hi * 16 + i) * TROW + nc;
                float v = (float)Tv[r] * (float)Tc[r] *
                          wnc[k0 + hi * 16 + i] * rinv;
                bb[i] = (__bf16)v;
            }
        }
#pragma unroll
        for (int mb = 0; mb < MBLK; ++mb) {
            union { uint4 u[2]; v16bf v; } A;
            const uint4* wr = (const uint4*)(w16 +
                (size_t)(m0 + mb * 16 + nc) * KK + k0 + hi * 16);
            if (s + 1 < STEPS) __builtin_prefetch(wr + 4, 0, 1);
            A.u[0] = wr[0];
            A.u[1] = wr[1];
            acc[mb] = __builtin_amdgcn_wmma_f32_16x16x32_bf16(
                false, A.v, false, bb, (short)0, acc[mb], false, false);
        }
    }
    float* ob = out + (size_t)b * M * HW;
#pragma unroll
    for (int mb = 0; mb < MBLK; ++mb)
#pragma unroll
        for (int r = 0; r < 8; ++r)
            ob[(size_t)(m0 + mb * 16 + hi * 8 + r) * HW + pos] = acc[mb][r];
}

// ---------------------------------------------------------------------------
extern "C" void kernel_launch(void* const* d_in, const int* in_sizes, int n_in,
                              void* d_out, int out_size, void* d_ws, size_t ws_size,
                              hipStream_t stream) {
    (void)in_sizes; (void)n_in; (void)out_size; (void)ws_size;
    const int B = 8, C = 128, H = 128, W = 128;
    const int HW = H * W, BHW = B * HW;
    const int C2 = 2 * C;   // 256
    const int C4 = 4 * C;   // 512

    const float* x        = (const float*)d_in[0];
    const float* evt      = (const float*)d_in[1];
    const float* w_n_img  = (const float*)d_in[2];
    const float* w_n_evt  = (const float*)d_in[3];
    const float* w_q      = (const float*)d_in[4];   // (256,128)
    const float* w_kv     = (const float*)d_in[5];   // (512,128)
    const float* w_q_dw   = (const float*)d_in[6];   // (256,9)
    const float* w_kv_dw  = (const float*)d_in[7];   // (512,9)
    const float* w_q_norm = (const float*)d_in[8];   // (256)
    const float* w_k_norm = (const float*)d_in[9];   // (256)
    const float* w_n_corr = (const float*)d_in[10];  // (256)
    const float* w_proj   = (const float*)d_in[11];  // (128,256)
    float* out = (float*)d_out;

    char* base = (char*)d_ws;
    size_t off = 0;
    auto take = [&](size_t bytes) -> char* {
        char* p = base + off;
        off = (off + bytes + 255) & ~(size_t)255;
        return p;
    };
    __bf16* wq16   = (__bf16*)take((size_t)C2 * C * 2);
    __bf16* wkv16  = (__bf16*)take((size_t)C4 * C * 2);
    __bf16* wpj16  = (__bf16*)take((size_t)C * C2 * 2);
    float*  rstd_x = (float*)take((size_t)BHW * 4);
    float*  rstd_e = (float*)take((size_t)BHW * 4);
    float*  rstd_q = (float*)take((size_t)BHW * 4);
    float*  rstd_k = (float*)take((size_t)BHW * 4);
    float*  rstd_c = (float*)take((size_t)BHW * 4);
    __bf16* q1     = (__bf16*)take((size_t)B * C2 * HW * 2);
    __bf16* kv1    = (__bf16*)take((size_t)B * C4 * HW * 2);
    __bf16* qdw    = (__bf16*)take((size_t)B * C2 * HW * 2);
    __bf16* kvdw   = (__bf16*)take((size_t)B * C4 * HW * 2);
    __bf16* corr   = (__bf16*)take((size_t)B * C2 * HW * 2);

    // 1. weights -> bf16
    cvt_f32_bf16_kernel<<<(C2 * C + 255) / 256, 256, 0, stream>>>(w_q, wq16, C2 * C);
    cvt_f32_bf16_kernel<<<(C4 * C + 255) / 256, 256, 0, stream>>>(w_kv, wkv16, C4 * C);
    cvt_f32_bf16_kernel<<<(C * C2 + 255) / 256, 256, 0, stream>>>(w_proj, wpj16, C * C2);

    // 2. rms stats for x, evt
    rms_stats_f32_kernel<<<(BHW + 255) / 256, 256, 0, stream>>>(x, rstd_x, C, HW, BHW);
    rms_stats_f32_kernel<<<(BHW + 255) / 256, 256, 0, stream>>>(evt, rstd_e, C, HW, BHW);

    // 3. conv1x1 GEMMs (WMMA bf16, async double-buffered B tiles)
    {
        dim3 gq(HW / 16, C2 / 64, B);
        gemm_conv1x1_bf16_kernel<128, 4><<<gq, 32, 0, stream>>>(
            x, wq16, w_n_img, rstd_x, q1, C2, HW);
        dim3 gkv(HW / 16, C4 / 64, B);
        gemm_conv1x1_bf16_kernel<128, 4><<<gkv, 32, 0, stream>>>(
            evt, wkv16, w_n_evt, rstd_e, kv1, C4, HW);
    }

    // 4. depthwise 3x3
    {
        int tq = B * C2 * HW;
        dwconv3x3_kernel<<<(tq + 255) / 256, 256, 0, stream>>>(q1, w_q_dw, qdw, C2, H, W, tq);
        int tkv = B * C4 * HW;
        dwconv3x3_kernel<<<(tkv + 255) / 256, 256, 0, stream>>>(kv1, w_kv_dw, kvdw, C4, H, W, tkv);
    }

    // 5. rms stats for q and k
    rms_stats_bf16_kernel<<<(BHW + 255) / 256, 256, 0, stream>>>(qdw, rstd_q, C2, 0, C2, HW, BHW);
    rms_stats_bf16_kernel<<<(BHW + 255) / 256, 256, 0, stream>>>(kvdw, rstd_k, C4, 0, C2, HW, BHW);

    // 6. patch circular conv
    {
        int nblk = B * (H / 8) * (W / 8);
        patch_corr_kernel<<<nblk, 256, 0, stream>>>(qdw, kvdw, w_q_norm, w_k_norm,
                                                    rstd_q, rstd_k, corr, C2, H, W);
    }

    // 7. rms stats for corr
    rms_stats_bf16_kernel<<<(BHW + 255) / 256, 256, 0, stream>>>(corr, rstd_c, C2, 0, C2, HW, BHW);

    // 8. final proj GEMM
    {
        dim3 gp(HW / 16, C / 64, B);
        gemm_vproj_kernel<256, 4><<<gp, 32, 0, stream>>>(
            kvdw, corr, wpj16, w_n_corr, rstd_c, out, C, HW);
    }
}
